// DGCNN_MidFusion_5274219839631
// MI455X (gfx1250) — matrix operand order repaired and checked by reading, not compiled, not run on hardware
//
#include <hip/hip_runtime.h>
#include <hip/hip_bf16.h>

// ---------------------------------------------------------------------------
// DGCNN mid-fusion forward for MI455X (gfx1250, wave32, WMMA).
// All matmuls use v_wmma_f32_16x16x32_f16 (f16 A/B, f32 accumulate). Every
// GEMM input is unit-scale and immediately BatchNorm'd, so f16 operands are
// numerically adequate. The kNN distance matrix is never materialized: a
// fused WMMA-Gram + register top-16 kernel streams column tiles through LDS.
// GEMM v2: 16x64 output strip per wave (4 WMMAs per A-fragment), weight slab
// staged in LDS via gfx1250 GLOBAL_LOAD_ASYNC_TO_LDS_B128 (ASYNCcnt).
// ---------------------------------------------------------------------------

typedef __attribute__((ext_vector_type(16))) _Float16 v16h;
typedef __attribute__((ext_vector_type(8)))  _Float16 v8h;
typedef __attribute__((ext_vector_type(8)))  float    v8f;

#if defined(__gfx1250__) && __has_builtin(__builtin_amdgcn_global_load_async_to_lds_b128) && __has_builtin(__builtin_amdgcn_s_wait_asynccnt)
#define HAVE_ASYNC_LDS 1
// exact parameter types per hipcc diagnostic: AS(1)/AS(3) pointers to
// GCC-vector int4 (vector_size(16)), non-const.
typedef int v4i_b128 __attribute__((vector_size(16)));
typedef __attribute__((address_space(1))) v4i_b128 g_v4i;
typedef __attribute__((address_space(3))) v4i_b128 l_v4i;
#else
#define HAVE_ASYNC_LDS 0
#endif

#define LRELU(v) ((v) > 0.f ? (v) : 0.2f * (v))

// ---- WMMA fragment loaders (contiguous k = 16*(l/16) + s for A and B, wave32) ----------
// A (16x32, f16): lane l -> row = l%16; half h = K[16*(l/16) + h].
__device__ __forceinline__ v16h load_a_frag(const _Float16* __restrict__ A,
                                            int lda, int m0, int kk, int lane) {
  const int row = m0 + (lane & 15);
  const int kb  = kk + ((lane >> 4) << 4);
  const _Float16* p = A + (size_t)row * lda + kb;
  v8h lo = *(const v8h*)(p);
  v8h hi = *(const v8h*)(p + 8);
  v16h a;
#pragma unroll
  for (int i = 0; i < 8; ++i) { a[i] = lo[i]; a[8 + i] = hi[i]; }
  return a;
}
// B (32x16, f16): lane l -> col = l%16; half h = K[16*(l/16) + h].
// Row-major weight matrix (O x Kd): col n of B = weight row n.
__device__ __forceinline__ v16h load_b_frag(const _Float16* __restrict__ W,
                                            int ldb, int n0, int kk, int lane) {
  const int col = n0 + (lane & 15);
  const int kb  = kk + ((lane >> 4) << 4);
  const _Float16* p = W + (size_t)col * ldb + kb;
  v8h lo = *(const v8h*)(p);
  v8h hi = *(const v8h*)(p + 8);
  v16h b;
#pragma unroll
  for (int i = 0; i < 8; ++i) { b[i] = lo[i]; b[8 + i] = hi[i]; }
  return b;
}

// ---- f32 -> f16 with zero channel padding ---------------------------------
__global__ void k_to_f16_pad(const float* __restrict__ S, _Float16* __restrict__ D,
                             int rows, int cin, int cpad) {
  int i = blockIdx.x * blockDim.x + threadIdx.x;
  if (i >= rows * cpad) return;
  int r = i / cpad, c = i - r * cpad;
  D[i] = (c < cin) ? (_Float16)S[(size_t)r * cin + c] : (_Float16)0.f;
}

// ---- per-point squared norm ------------------------------------------------
__global__ void k_sqnorm(const float* __restrict__ X, float* __restrict__ XX,
                         int rows, int C) {
  int i = blockIdx.x * blockDim.x + threadIdx.x;
  if (i >= rows) return;
  const float* p = X + (size_t)i * C;
  float s = 0.f;
  for (int c = 0; c < C; ++c) s += p[c] * p[c];
  XX[i] = s;
}

// ---- fused WMMA Gram + top-16 neighbor selection ---------------------------
// One wave (block of 32) per 16-row tile of one batch. Sweeps column tiles,
// Gram via v_wmma, -||xi-xj||^2 tile spilled to LDS, sorted top-16 insertion
// per row (lanes 0..15 own one row each).
__global__ void k_dist_topk(const _Float16* __restrict__ Xh, const float* __restrict__ XX,
                            int* __restrict__ idxOut, int Npts, int Cp) {
  __shared__ float tile[16][17];
  const int b    = blockIdx.y;
  const int n0   = blockIdx.x << 4;
  const int lane = threadIdx.x;
  const _Float16* Xb = Xh + (size_t)b * Npts * Cp;
  const float*   xxb = XX + (size_t)b * Npts;

  float vals[16]; int idxs[16];
#pragma unroll
  for (int i = 0; i < 16; ++i) { vals[i] = -3.4e38f; idxs[i] = 0; }

  const float xrow = (lane < 16) ? xxb[n0 + lane] : 0.f;
  const int r0 = (lane >> 4) << 3;   // C-frag row base: 0 or 8
  const int cl = lane & 15;          // C-frag column

  for (int m0 = 0; m0 < Npts; m0 += 16) {
    v8f acc = {};
    for (int kk = 0; kk < Cp; kk += 32) {
      v16h af = load_a_frag(Xb, Cp, n0, kk, lane);
      v16h bf = load_b_frag(Xb, Cp, m0, kk, lane);
      acc = __builtin_amdgcn_wmma_f32_16x16x32_f16(false, af, false, bf,
                                                   (short)0, acc, false, false);
    }
    const float xcol = xxb[m0 + cl];
#pragma unroll
    for (int r = 0; r < 8; ++r) tile[r0 + r][cl] = 2.f * acc[r] - xcol;
    __syncthreads();
    if (lane < 16) {
      for (int c = 0; c < 16; ++c) {
        float v = tile[lane][c] - xrow;          // -||xi - xj||^2
        if (v > vals[15]) {                      // insert into sorted top-16
          vals[15] = v; idxs[15] = m0 + c;
#pragma unroll
          for (int i = 15; i > 0; --i) {
            if (vals[i] > vals[i - 1]) {
              float tv = vals[i]; vals[i] = vals[i - 1]; vals[i - 1] = tv;
              int   ti = idxs[i]; idxs[i] = idxs[i - 1]; idxs[i - 1] = ti;
            }
          }
        }
      }
    }
    __syncthreads();
  }
  if (lane < 16) {
    int* op = idxOut + ((size_t)(b * Npts + n0 + lane)) * 16;
#pragma unroll
    for (int i = 0; i < 16; ++i) op[i] = idxs[i];
  }
}

// ---- edge feature build: [nbr - ctr, ctr] in f16, K-padded -----------------
__global__ void k_build_edge(const float* __restrict__ X, const int* __restrict__ idx,
                             _Float16* __restrict__ E, int Npts, int C, int Kp,
                             int totalRows) {
  int row = blockIdx.x * blockDim.x + threadIdx.x;   // (b*N + n)*16 + k
  if (row >= totalRows) return;
  int bn = row >> 4;
  int b  = bn / Npts;
  int j  = idx[row];
  const float* ctr = X + (size_t)bn * C;
  const float* nbr = X + ((size_t)(b * Npts + j)) * C;
  _Float16* e = E + (size_t)row * Kp;
  for (int c = 0; c < C; ++c) {
    float cv = ctr[c];
    e[c]     = (_Float16)(nbr[c] - cv);
    e[C + c] = (_Float16)cv;
  }
  for (int c = 2 * C; c < Kp; ++c) e[c] = (_Float16)0.f;
}

// ---- WMMA GEMM v2: C[M,O] = A[M,Kd](f16) * W[O,Kd]^T(f16) ------------------
// Block = 4 waves, one 64-wide output slab (blockIdx.y). The 64 x Kd weight
// slab is staged to LDS once per block (async-to-LDS when available); each
// wave then computes a 16x64 strip: 1 A-fragment feeds 4 WMMAs per K-step.
#define GEMM_BN   64
#define GEMM_MAXK 768

__global__ __launch_bounds__(128)
void k_gemm_wmma(const _Float16* __restrict__ A, const _Float16* __restrict__ W,
                 float* __restrict__ C, int M, int Nout, int Kd) {
  __shared__ _Float16 sW[GEMM_BN * GEMM_MAXK];
  const int tid  = threadIdx.x;
  const int lane = tid & 31;
  const int wv   = tid >> 5;
  const int nb   = blockIdx.y << 6;                 // output-column slab base

  // stage 64 x Kd weight slab into LDS (coalesced 16B transfers)
  const _Float16* Wslab = W + (size_t)nb * Kd;
  const int halves = GEMM_BN * Kd;
#if HAVE_ASYNC_LDS
  for (int i = tid * 8; i < halves; i += 128 * 8)
    __builtin_amdgcn_global_load_async_to_lds_b128((g_v4i*)(Wslab + i),
                                                   (l_v4i*)(sW + i), 0, 0);
  __builtin_amdgcn_s_wait_asynccnt(0);
#else
  for (int i = tid * 8; i < halves; i += 128 * 8)
    *(v8h*)(sW + i) = *(const v8h*)(Wslab + i);
#endif
  __syncthreads();

  const int mt = blockIdx.x * 4 + wv;               // wave-uniform
  if (mt < (M >> 4)) {
    const int m0 = mt << 4;
    v8f acc0 = {}, acc1 = {}, acc2 = {}, acc3 = {};
    for (int kk = 0; kk < Kd; kk += 32) {
      if (kk + 32 < Kd)                             // global_prefetch_b8
        __builtin_prefetch((const void*)(A + (size_t)(m0 + (lane & 15)) * Kd + kk + 32), 0, 1);
      v16h af = load_a_frag(A, Kd, m0, kk, lane);
      v16h b0 = load_b_frag(sW, Kd, 0,  kk, lane);  // LDS-resident B frags
      v16h b1 = load_b_frag(sW, Kd, 16, kk, lane);
      v16h b2 = load_b_frag(sW, Kd, 32, kk, lane);
      v16h b3 = load_b_frag(sW, Kd, 48, kk, lane);
      acc0 = __builtin_amdgcn_wmma_f32_16x16x32_f16(false, af, false, b0, (short)0, acc0, false, false);
      acc1 = __builtin_amdgcn_wmma_f32_16x16x32_f16(false, af, false, b1, (short)0, acc1, false, false);
      acc2 = __builtin_amdgcn_wmma_f32_16x16x32_f16(false, af, false, b2, (short)0, acc2, false, false);
      acc3 = __builtin_amdgcn_wmma_f32_16x16x32_f16(false, af, false, b3, (short)0, acc3, false, false);
    }
    const int row0 = m0 + ((lane >> 4) << 3);
    const int cl   = lane & 15;
    float* Cp0 = C + (size_t)row0 * Nout + nb + cl;
#pragma unroll
    for (int r = 0; r < 8; ++r) {
      float* cr = Cp0 + (size_t)r * Nout;
      cr[0]  = acc0[r];
      cr[16] = acc1[r];
      cr[32] = acc2[r];
      cr[48] = acc3[r];
    }
  }
}

// ---- train-mode BN statistics: one block per output channel ----------------
// Column-strided reads are L2-resident (hbuf <= 134 MB < 192 MB L2).
__global__ void k_bn_stats(const float* __restrict__ H, float* __restrict__ mean,
                           float* __restrict__ rstd, int M, int O) {
  const int o = blockIdx.x, t = threadIdx.x;
  float s = 0.f, s2 = 0.f;
  for (int m = t; m < M; m += blockDim.x) {
    float v = H[(size_t)m * O + o]; s += v; s2 += v * v;
  }
  __shared__ float sh[256], sh2[256];
  sh[t] = s; sh2[t] = s2; __syncthreads();
  for (int w = 128; w > 0; w >>= 1) {
    if (t < w) { sh[t] += sh[t + w]; sh2[t] += sh2[t + w]; }
    __syncthreads();
  }
  if (t == 0) {
    float mu  = sh[0] / (float)M;
    float var = sh2[0] / (float)M - mu * mu;
    mean[o] = mu; rstd[o] = rsqrtf(var + 1e-5f);
  }
}

// ---- normalize + LeakyReLU + max over K=16 (EdgeConv epilogue) -------------
__global__ void k_edge_reduce(const float* __restrict__ H, const float* __restrict__ mean,
                              const float* __restrict__ rstd, float* __restrict__ Xout,
                              int rows, int O) {
  int i = blockIdx.x * blockDim.x + threadIdx.x;
  if (i >= rows * O) return;
  int bn = i / O, o = i - bn * O;
  float mu = mean[o], r = rstd[o];
  const float* hp = H + (size_t)bn * 16 * O + o;
  float mx = -3.4e38f;
#pragma unroll
  for (int k = 0; k < 16; ++k) {
    float v = (hp[(size_t)k * O] - mu) * r;
    mx = fmaxf(mx, LRELU(v));
  }
  Xout[(size_t)bn * O + o] = mx;
}

// ---- normalize + LeakyReLU (conv_bn epilogue); optional f32/f16 outputs ----
__global__ void k_bn_act(const float* __restrict__ H, const float* __restrict__ mean,
                         const float* __restrict__ rstd, float* __restrict__ Yf,
                         _Float16* __restrict__ Yh, int M, int O) {
  int i = blockIdx.x * blockDim.x + threadIdx.x;
  if (i >= M * O) return;
  int o = i % O;
  float v = (H[i] - mean[o]) * rstd[o];
  v = LRELU(v);
  if (Yf) Yf[i] = v;
  if (Yh) Yh[i] = (_Float16)v;
}

// ---- normalize + LeakyReLU + global max-pool over N ------------------------
__global__ void k_bn_act_maxpool(const float* __restrict__ H, const float* __restrict__ mean,
                                 const float* __restrict__ rstd, float* __restrict__ G,
                                 int Npts, int O, int Bb) {
  int i = blockIdx.x * blockDim.x + threadIdx.x;
  if (i >= Bb * O) return;
  int b = i / O, o = i - b * O;
  float mu = mean[o], r = rstd[o];
  const float* hp = H + ((size_t)b * Npts) * O + o;
  float mx = -3.4e38f;
  for (int n = 0; n < Npts; ++n) {
    float v = (hp[(size_t)n * O] - mu) * r;
    mx = fmaxf(mx, LRELU(v));
  }
  G[i] = mx;
}

// ---- channel concat (two f32 sources) into f16 -----------------------------
__global__ void k_concat2(const float* __restrict__ A, const float* __restrict__ Bs,
                          _Float16* __restrict__ D, int M, int Ca, int Cb) {
  int i = blockIdx.x * blockDim.x + threadIdx.x;
  int Ct = Ca + Cb;
  if (i >= M * Ct) return;
  int m = i / Ct, c = i - m * Ct;
  float v = (c < Ca) ? A[(size_t)m * Ca + c] : Bs[(size_t)m * Cb + (c - Ca)];
  D[i] = (_Float16)v;
}

// ---- concat per-point features with per-batch broadcast vector -------------
__global__ void k_concat_bcast(const float* __restrict__ F, const float* __restrict__ G,
                               _Float16* __restrict__ D, int Npts, int Ca, int Cb, int M) {
  int i = blockIdx.x * blockDim.x + threadIdx.x;
  int Ct = Ca + Cb;
  if (i >= M * Ct) return;
  int m = i / Ct, c = i - m * Ct;
  float v = (c < Ca) ? F[(size_t)m * Ca + c]
                     : G[(size_t)(m / Npts) * Cb + (c - Ca)];
  D[i] = (_Float16)v;
}

// ---- final 6-channel head + bias + (B,N,6)->(B,6,N) transpose --------------
__global__ void k_logits(const float* __restrict__ H2, const float* __restrict__ W3,
                         const float* __restrict__ b3, float* __restrict__ Out,
                         int Npts, int Bb) {
  int i = blockIdx.x * blockDim.x + threadIdx.x;
  if (i >= Bb * Npts * 6) return;
  int o  = i % 6;
  int bn = i / 6;
  int b  = bn / Npts, n = bn - b * Npts;
  const float* hp = H2 + (size_t)bn * 128;
  const float* wp = W3 + (size_t)o * 128;
  float s = b3[o];
  for (int c = 0; c < 128; ++c) s += hp[c] * wp[c];
  Out[((size_t)b * 6 + o) * Npts + n] = s;
}

// ---------------------------------------------------------------------------
extern "C" void kernel_launch(void* const* d_in, const int* in_sizes, int n_in,
                              void* d_out, int out_size, void* d_ws, size_t ws_size,
                              hipStream_t stream) {
  (void)in_sizes; (void)n_in; (void)out_size; (void)ws_size;
  const float* spatial  = (const float*)d_in[0];
  const float* spectral = (const float*)d_in[1];
  const float* W1  = (const float*)d_in[2];
  const float* W2  = (const float*)d_in[3];
  const float* W3  = (const float*)d_in[4];
  const float* W4  = (const float*)d_in[5];
  const float* V1  = (const float*)d_in[6];
  const float* V2  = (const float*)d_in[7];
  const float* V3  = (const float*)d_in[8];
  const float* V4  = (const float*)d_in[9];
  const float* Wf  = (const float*)d_in[10];
  const float* Wg  = (const float*)d_in[11];
  const float* Wh1 = (const float*)d_in[12];
  const float* Wh2 = (const float*)d_in[13];
  const float* Wh3 = (const float*)d_in[14];
  const float* bh3 = (const float*)d_in[15];
  float* Out = (float*)d_out;

  const int B = 8, Npts = 2048;
  const int Mpt = B * Npts;     // 16384 points
  const int Me  = Mpt * 16;     // 262144 edge rows

  // bump allocator over workspace (assumes generous ws; peak ~400 MB)
  char* base = (char*)d_ws;
  size_t off = 0;
  auto alloc = [&](size_t bytes) -> void* {
    off = (off + 255) & ~(size_t)255;
    void* p = base + off; off += bytes; return p;
  };

  _Float16* wW1h  = (_Float16*)alloc((size_t)64  * 32  * 2);
  _Float16* wW2h  = (_Float16*)alloc((size_t)64  * 128 * 2);
  _Float16* wW3h  = (_Float16*)alloc((size_t)128 * 128 * 2);
  _Float16* wW4h  = (_Float16*)alloc((size_t)128 * 256 * 2);
  _Float16* wV1h  = (_Float16*)alloc((size_t)64  * 32  * 2);
  _Float16* wV2h  = (_Float16*)alloc((size_t)64  * 64  * 2);
  _Float16* wV3h  = (_Float16*)alloc((size_t)128 * 64  * 2);
  _Float16* wV4h  = (_Float16*)alloc((size_t)128 * 128 * 2);
  _Float16* wWfh  = (_Float16*)alloc((size_t)256 * 256 * 2);
  _Float16* wWgh  = (_Float16*)alloc((size_t)512 * 256 * 2);
  _Float16* wWh1h = (_Float16*)alloc((size_t)256 * 768 * 2);
  _Float16* wWh2h = (_Float16*)alloc((size_t)128 * 256 * 2);

  _Float16* xh      = (_Float16*)alloc((size_t)Mpt * 128 * 2);
  float*    xx      = (float*)   alloc((size_t)Mpt * 4);
  int*      idxb    = (int*)     alloc((size_t)Me * 4);
  _Float16* edge    = (_Float16*)alloc((size_t)Me * 256 * 2);
  float*    hbuf    = (float*)   alloc((size_t)Me * 128 * 4);
  float*    xA      = (float*)   alloc((size_t)Mpt * 128 * 4);
  float*    xB      = (float*)   alloc((size_t)Mpt * 128 * 4);
  _Float16* s0h     = (_Float16*)alloc((size_t)Mpt * 32  * 2);
  float*    s1f     = (float*)   alloc((size_t)Mpt * 64  * 4);
  _Float16* s1h     = (_Float16*)alloc((size_t)Mpt * 64  * 2);
  float*    s2f     = (float*)   alloc((size_t)Mpt * 64  * 4);
  _Float16* s2h     = (_Float16*)alloc((size_t)Mpt * 64  * 2);
  float*    s3f     = (float*)   alloc((size_t)Mpt * 128 * 4);
  _Float16* s3h     = (_Float16*)alloc((size_t)Mpt * 128 * 2);
  float*    s4f     = (float*)   alloc((size_t)Mpt * 128 * 4);
  _Float16* s4h     = (_Float16*)alloc((size_t)Mpt * 128 * 2);
  _Float16* cat256h = (_Float16*)alloc((size_t)Mpt * 256 * 2);
  float*    fusedF  = (float*)   alloc((size_t)Mpt * 256 * 4);
  _Float16* fusedH  = (_Float16*)alloc((size_t)Mpt * 256 * 2);
  float*    gbuf    = (float*)   alloc((size_t)B * 512 * 4);
  _Float16* cat768h = (_Float16*)alloc((size_t)Mpt * 768 * 2);
  _Float16* h1h     = (_Float16*)alloc((size_t)Mpt * 256 * 2);
  float*    h2f     = (float*)   alloc((size_t)Mpt * 128 * 4);
  float*    meanb   = (float*)   alloc(512 * 4);
  float*    rstdb   = (float*)   alloc(512 * 4);

  auto nblk = [](long long tot, int bs) { return (unsigned)((tot + bs - 1) / bs); };

  auto cvt = [&](const float* src, _Float16* dst, int rows, int cin, int cpad) {
    k_to_f16_pad<<<nblk((long long)rows * cpad, 256), 256, 0, stream>>>(src, dst, rows, cin, cpad);
  };
  auto gemm = [&](const _Float16* A, const _Float16* Wm, float* Cc, int M, int O, int Kd) {
    dim3 g((unsigned)(((M / 16) + 3) / 4), (unsigned)(O / 64));
    k_gemm_wmma<<<g, 128, 0, stream>>>(A, Wm, Cc, M, O, Kd);
  };
  auto conv_layer = [&](const _Float16* Ah, int Cp, const _Float16* Wm, int O,
                        float* Yf, _Float16* Yh, int M) {
    gemm(Ah, Wm, hbuf, M, O, Cp);
    k_bn_stats<<<O, 256, 0, stream>>>(hbuf, meanb, rstdb, M, O);
    k_bn_act<<<nblk((long long)M * O, 256), 256, 0, stream>>>(hbuf, meanb, rstdb, Yf, Yh, M, O);
  };
  auto edge_layer = [&](const float* Xin, int C, int Cp, const _Float16* Wm, int O, float* Xout) {
    cvt(Xin, xh, Mpt, C, Cp);
    k_sqnorm<<<nblk(Mpt, 256), 256, 0, stream>>>(Xin, xx, Mpt, C);
    k_dist_topk<<<dim3(Npts / 16, B), 32, 0, stream>>>(xh, xx, idxb, Npts, Cp);
    const int Kp = (2 * C < 32) ? 32 : 2 * C;
    k_build_edge<<<nblk(Me, 256), 256, 0, stream>>>(Xin, idxb, edge, Npts, C, Kp, Me);
    gemm(edge, Wm, hbuf, Me, O, Kp);
    k_bn_stats<<<O, 256, 0, stream>>>(hbuf, meanb, rstdb, Me, O);
    k_edge_reduce<<<nblk((long long)Mpt * O, 256), 256, 0, stream>>>(hbuf, meanb, rstdb, Xout, Mpt, O);
  };

  // weight conversion (zero K-padding to multiples of 32 where needed)
  cvt(W1, wW1h, 64, 10, 32);      cvt(W2, wW2h, 64, 128, 128);
  cvt(W3, wW3h, 128, 128, 128);   cvt(W4, wW4h, 128, 256, 256);
  cvt(V1, wV1h, 64, 5, 32);       cvt(V2, wV2h, 64, 64, 64);
  cvt(V3, wV3h, 128, 64, 64);     cvt(V4, wV4h, 128, 128, 128);
  cvt(Wf, wWfh, 256, 256, 256);   cvt(Wg, wWgh, 512, 256, 256);
  cvt(Wh1, wWh1h, 256, 768, 768); cvt(Wh2, wWh2h, 128, 256, 256);

  // EdgeConv chain (spatial branch)
  edge_layer(spatial, 5, 32,   wW1h, 64,  xA);
  edge_layer(xA,      64, 64,  wW2h, 64,  xB);
  edge_layer(xB,      64, 64,  wW3h, 128, xA);
  edge_layer(xA,     128, 128, wW4h, 128, xB);

  // spectral MLP chain
  cvt(spectral, s0h, Mpt, 5, 32);
  conv_layer(s0h, 32,  wV1h, 64,  s1f, s1h, Mpt);
  conv_layer(s1h, 64,  wV2h, 64,  s2f, s2h, Mpt);
  conv_layer(s2h, 64,  wV3h, 128, s3f, s3h, Mpt);
  conv_layer(s3h, 128, wV4h, 128, s4f, s4h, Mpt);

  // fusion
  k_concat2<<<nblk((long long)Mpt * 256, 256), 256, 0, stream>>>(xB, s4f, cat256h, Mpt, 128, 128);
  conv_layer(cat256h, 256, wWfh, 256, fusedF, fusedH, Mpt);

  // global branch: conv_bn(Wg) then max over N, fused into the epilogue
  gemm(fusedH, wWgh, hbuf, Mpt, 512, 256);
  k_bn_stats<<<512, 256, 0, stream>>>(hbuf, meanb, rstdb, Mpt, 512);
  k_bn_act_maxpool<<<nblk((long long)B * 512, 256), 256, 0, stream>>>(hbuf, meanb, rstdb, gbuf, Npts, 512, B);

  // head
  k_concat_bcast<<<nblk((long long)Mpt * 768, 256), 256, 0, stream>>>(fusedF, gbuf, cat768h, Npts, 256, 512, Mpt);
  conv_layer(cat768h, 768, wWh1h, 256, (float*)nullptr, h1h, Mpt);
  conv_layer(h1h, 256, wWh2h, 128, h2f, (_Float16*)nullptr, Mpt);
  k_logits<<<nblk((long long)Mpt * 6, 256), 256, 0, stream>>>(h2f, Wh3, bh3, Out, Npts, B);
}
